// BiochemicalConstraintLayer_9509057593392
// MI455X (gfx1250) — matrix operand order repaired and checked by reading, not compiled, not run on hardware
//
#include <hip/hip_runtime.h>
#include <hip/hip_bf16.h>

typedef __attribute__((ext_vector_type(16))) _Float16 v16h;
typedef __attribute__((ext_vector_type(8)))  _Float16 v8h;
typedef __attribute__((ext_vector_type(8)))  float    v8f;

#define D_FEAT 128

// ---- A-fragment loaders: 8 consecutive K values into elements [B..B+7] ----
template<int B>
__device__ __forceinline__ void load8(const float* __restrict__ p, v16h& a) {
    float4 x = *(const float4*)p;
    float4 y = *(const float4*)(p + 4);
    a[B + 0] = (_Float16)x.x; a[B + 1] = (_Float16)x.y;
    a[B + 2] = (_Float16)x.z; a[B + 3] = (_Float16)x.w;
    a[B + 4] = (_Float16)y.x; a[B + 5] = (_Float16)y.y;
    a[B + 6] = (_Float16)y.z; a[B + 7] = (_Float16)y.w;
}

template<int B>
__device__ __forceinline__ void load8(const _Float16* __restrict__ p, v16h& a) {
    v8h x = *(const v8h*)p;          // single 16-byte load, no conversion
    #pragma unroll
    for (int i = 0; i < 8; ++i) a[B + i] = x[i];
}

// ---- one-time fp32 -> fp16 staging of node features (8 elems / thread) ----
__global__ __launch_bounds__(256) void cvt_f16_kernel(const float* __restrict__ in,
                                                      _Float16* __restrict__ out,
                                                      long n8) {
    long i = (long)blockIdx.x * blockDim.x + threadIdx.x;
    if (i >= n8) return;
    const float4* p = (const float4*)(in + i * 8);
    float4 x = p[0], y = p[1];
    v8h o;
    o[0] = (_Float16)x.x; o[1] = (_Float16)x.y; o[2] = (_Float16)x.z; o[3] = (_Float16)x.w;
    o[4] = (_Float16)y.x; o[5] = (_Float16)y.y; o[6] = (_Float16)y.z; o[7] = (_Float16)y.w;
    *(v8h*)(out + i * 8) = o;
}

// Pack fp32 row-major W[K,Nn] into per-lane WMMA B-fragment layout:
// out[((kt*NT + nt)*32 + lane)*16 + e] = (f16) W[(kt*32 + 8*(lane/16) + (e%8) + 16*(e/8))*Nn + nt*16 + lane%16]
__global__ void pack_b_kernel(const float* __restrict__ W, _Float16* __restrict__ out,
                              int K, int Nn) {
    int idx = blockIdx.x * blockDim.x + threadIdx.x;
    int NT = Nn / 16;
    int total = (K / 32) * NT * 32 * 16;
    if (idx >= total) return;
    int e   = idx & 15;
    int l   = (idx >> 4) & 31;
    int blk = idx >> 9;
    int kt  = blk / NT;
    int nt  = blk % NT;
    int k   = kt * 32 + (l >> 4) * 8 + (e & 7) + (e >> 3) * 16;
    int n   = nt * 16 + (l & 15);
    out[idx] = (_Float16)W[k * Nn + n];
}

// Valence MLP: per wave, 16 nodes. WMMA 128->32, then scalar 32->8 + softmax + argmax.
template<typename T>
__global__ __launch_bounds__(256) void valence_kernel(
    const T* __restrict__ nf, const _Float16* __restrict__ w1p,
    const float* __restrict__ vb1, const float* __restrict__ vw2,
    const float* __restrict__ vb2, float* __restrict__ val_out,
    float* __restrict__ pred, int nTiles, int N) {
    __shared__ float hid[8][16][32];
    const int tid  = threadIdx.x;
    const int wave = tid >> 5;
    const int lane = tid & 31;
    const int lrow = lane & 15;
    const int half = lane >> 4;
    const int tile = blockIdx.x * 8 + wave;
    const bool active = (tile < nTiles);

    if (active) {
        v8f c0 = {}; v8f c1 = {};
        long node = (long)tile * 16 + lrow;
        if (node > N - 1) node = N - 1;
        const T* fp = nf + node * D_FEAT;
        #pragma unroll
        for (int kt = 0; kt < 4; ++kt) {
            v16h a;
            int kb = kt * 32 + half * 8;
            load8<0>(fp + kb, a);
            load8<8>(fp + kb + 16, a);
            v16h b0 = *(const v16h*)(w1p + ((kt * 2 + 0) * 32 + lane) * 16);
            v16h b1 = *(const v16h*)(w1p + ((kt * 2 + 1) * 32 + lane) * 16);
            c0 = __builtin_amdgcn_wmma_f32_16x16x32_f16(false, a, false, b0, (short)0, c0, false, false);
            c1 = __builtin_amdgcn_wmma_f32_16x16x32_f16(false, a, false, b1, (short)0, c1, false, false);
        }
        float bias0 = vb1[lrow];
        float bias1 = vb1[16 + lrow];
        #pragma unroll
        for (int v = 0; v < 8; ++v) {
            int r = v + half * 8;
            float h0 = c0[v] + bias0; h0 = h0 > 0.f ? h0 : 0.f;
            float h1 = c1[v] + bias1; h1 = h1 > 0.f ? h1 : 0.f;
            hid[wave][r][lrow]      = h0;
            hid[wave][r][16 + lrow] = h1;
        }
    }
    __syncthreads();
    if (active && lane < 16) {
        int node = tile * 16 + lane;
        if (node < N) {
            float o[8];
            #pragma unroll
            for (int j = 0; j < 8; ++j) o[j] = vb2[j];
            for (int k = 0; k < 32; ++k) {
                float h = hid[wave][lane][k];
                #pragma unroll
                for (int j = 0; j < 8; ++j) o[j] += h * vw2[k * 8 + j];
            }
            float mx = o[0]; int am = 0;
            #pragma unroll
            for (int j = 1; j < 8; ++j) if (o[j] > mx) { mx = o[j]; am = j; }
            float s = 0.f;
            #pragma unroll
            for (int j = 0; j < 8; ++j) { o[j] = __expf(o[j] - mx); s += o[j]; }
            float inv = 1.0f / s;
            float* dst = val_out + (long)node * 8;
            #pragma unroll
            for (int j = 0; j < 8; ++j) dst[j] = o[j] * inv;
            pred[node] = (float)(am + 1);
        }
    }
}

// Edge MLP: per wave, 16 edges. Gathered [16,256] @ [256,32] via WMMA, then 32->4 + softmax,
// bond order and scatter-add to node degree.
template<typename T>
__global__ __launch_bounds__(256) void edge_kernel(
    const T* __restrict__ nf, const long long* __restrict__ ei,
    const _Float16* __restrict__ w1p, const float* __restrict__ bb1,
    const float* __restrict__ bw2, const float* __restrict__ bb2,
    float* __restrict__ bond_out, float* __restrict__ deg,
    long eTiles, long E) {
    __shared__ float hid[8][16][32];
    const int tid  = threadIdx.x;
    const int wave = tid >> 5;
    const int lane = tid & 31;
    const int lrow = lane & 15;
    const int half = lane >> 4;
    const long tile = (long)blockIdx.x * 8 + wave;
    const bool active = (tile < eTiles);
    long rown = 0;

    if (active) {
        long e = tile * 16 + lrow;
        if (e > E - 1) e = E - 1;
        rown      = ei[e];
        long coln = ei[E + e];
        const T* rp = nf + rown * D_FEAT;
        const T* cp = nf + coln * D_FEAT;
        v8f c0 = {}; v8f c1 = {};
        #pragma unroll
        for (int kt = 0; kt < 8; ++kt) {
            const T* fp = (kt < 4) ? rp : cp;      // K<128 -> row endpoint, else col
            int kb = (kt * 32 + half * 8) & 127;
            v16h a;
            load8<0>(fp + kb, a);
            load8<8>(fp + kb + 16, a);
            v16h b0 = *(const v16h*)(w1p + ((kt * 2 + 0) * 32 + lane) * 16);
            v16h b1 = *(const v16h*)(w1p + ((kt * 2 + 1) * 32 + lane) * 16);
            c0 = __builtin_amdgcn_wmma_f32_16x16x32_f16(false, a, false, b0, (short)0, c0, false, false);
            c1 = __builtin_amdgcn_wmma_f32_16x16x32_f16(false, a, false, b1, (short)0, c1, false, false);
        }
        float bias0 = bb1[lrow];
        float bias1 = bb1[16 + lrow];
        #pragma unroll
        for (int v = 0; v < 8; ++v) {
            int r = v + half * 8;
            float h0 = c0[v] + bias0; h0 = h0 > 0.f ? h0 : 0.f;
            float h1 = c1[v] + bias1; h1 = h1 > 0.f ? h1 : 0.f;
            hid[wave][r][lrow]      = h0;
            hid[wave][r][16 + lrow] = h1;
        }
    }
    __syncthreads();
    if (active && lane < 16) {
        long e = tile * 16 + lane;
        if (e < E) {
            float o[4];
            #pragma unroll
            for (int j = 0; j < 4; ++j) o[j] = bb2[j];
            for (int k = 0; k < 32; ++k) {
                float h = hid[wave][lane][k];
                #pragma unroll
                for (int j = 0; j < 4; ++j) o[j] += h * bw2[k * 4 + j];
            }
            float mx = o[0];
            #pragma unroll
            for (int j = 1; j < 4; ++j) mx = o[j] > mx ? o[j] : mx;
            float s = 0.f;
            #pragma unroll
            for (int j = 0; j < 4; ++j) { o[j] = __expf(o[j] - mx); s += o[j]; }
            float inv = 1.0f / s;
            float* dst = bond_out + e * 4;
            #pragma unroll
            for (int j = 0; j < 4; ++j) dst[j] = o[j] * inv;
            float bond = o[0] * inv * 1.0f + o[1] * inv * 2.0f +
                         o[2] * inv * 3.0f + o[3] * inv * 1.5f;
            unsafeAtomicAdd(&deg[rown], bond);
        }
    }
}

__global__ __launch_bounds__(256) void finalize_kernel(
    const float* __restrict__ deg, const float* __restrict__ pred,
    float* __restrict__ out0, int N) {
    __shared__ float red[256];
    int i = blockIdx.x * 256 + threadIdx.x;
    float v = 0.f;
    if (i < N) { float d = deg[i] - pred[i]; v = d * d; }
    red[threadIdx.x] = v;
    __syncthreads();
    #pragma unroll
    for (int s = 128; s > 0; s >>= 1) {
        if (threadIdx.x < s) red[threadIdx.x] += red[threadIdx.x + s];
        __syncthreads();
    }
    if (threadIdx.x == 0) unsafeAtomicAdd(out0, red[0] * (1.0f / (float)N));
}

extern "C" void kernel_launch(void* const* d_in, const int* in_sizes, int n_in,
                              void* d_out, int out_size, void* d_ws, size_t ws_size,
                              hipStream_t stream) {
    const float*     nf  = (const float*)d_in[0];
    const long long* ei  = (const long long*)d_in[1];   // int64 [2, E]
    const float*     vw1 = (const float*)d_in[2];
    const float*     vb1 = (const float*)d_in[3];
    const float*     vw2 = (const float*)d_in[4];
    const float*     vb2 = (const float*)d_in[5];
    const float*     bw1 = (const float*)d_in[6];
    const float*     bb1 = (const float*)d_in[7];
    const float*     bw2 = (const float*)d_in[8];
    const float*     bb2 = (const float*)d_in[9];

    const int  N = in_sizes[0] / D_FEAT;
    const long E = (long)in_sizes[1] / 2;
    float* out = (float*)d_out;

    // Workspace layout (ws base 256B aligned; all sub-offsets keep >=16B alignment):
    char*      ws    = (char*)d_ws;
    float*     pred  = (float*)ws;                                   // N floats
    float*     deg   = (float*)(ws + (size_t)N * 4);                 // N floats
    _Float16*  vw1p  = (_Float16*)(ws + (size_t)N * 8);              // 4096 f16 (8 KB)
    _Float16*  bw1p  = (_Float16*)(ws + (size_t)N * 8 + 4096 * 2);   // 8192 f16 (16 KB)
    size_t     base  = (size_t)N * 8 + 4096 * 2 + 8192 * 2;
    _Float16*  nfh   = (_Float16*)(ws + base);                       // optional: N*128 f16
    const size_t needF16 = base + (size_t)N * D_FEAT * 2;
    const bool useF16 = (ws_size >= needF16);    // ws_size constant across calls -> deterministic

    hipMemsetAsync(deg, 0, (size_t)N * 4, stream);
    hipMemsetAsync(out, 0, sizeof(float), stream);

    pack_b_kernel<<<(4096 + 255) / 256, 256, 0, stream>>>(vw1, vw1p, 128, 32);
    pack_b_kernel<<<(8192 + 255) / 256, 256, 0, stream>>>(bw1, bw1p, 256, 32);

    const int  nTiles  = (N + 15) / 16;
    const int  blocksA = (nTiles + 7) / 8;
    const long eTiles  = (E + 15) / 16;
    const int  blocksB = (int)((eTiles + 7) / 8);
    float*     valOut  = out + 1;
    float*     bondOut = out + 1 + (long)N * 8;

    if (useF16) {
        // Stage node features to fp16 once: halves L2 gather traffic and removes
        // all per-tile cvt VALU work from the WMMA hot loop.
        long n8 = (long)N * D_FEAT / 8;
        cvt_f16_kernel<<<(int)((n8 + 255) / 256), 256, 0, stream>>>(nf, nfh, n8);
        valence_kernel<_Float16><<<blocksA, 256, 0, stream>>>(nfh, vw1p, vb1, vw2, vb2,
                                                              valOut, pred, nTiles, N);
        edge_kernel<_Float16><<<blocksB, 256, 0, stream>>>(nfh, ei, bw1p, bb1, bw2, bb2,
                                                           bondOut, deg, eTiles, E);
    } else {
        valence_kernel<float><<<blocksA, 256, 0, stream>>>(nf, vw1p, vb1, vw2, vb2,
                                                           valOut, pred, nTiles, N);
        edge_kernel<float><<<blocksB, 256, 0, stream>>>(nf, ei, bw1p, bb1, bw2, bb2,
                                                        bondOut, deg, eTiles, E);
    }

    finalize_kernel<<<(N + 255) / 256, 256, 0, stream>>>(deg, pred, out, N);
}
